// NPerTokenSwishGLU_Basis_64613488001347
// MI455X (gfx1250) — matrix operand order repaired and checked by reading, not compiled, not run on hardware
//
#include <hip/hip_runtime.h>
#include <hip/hip_fp16.h>
#include <math.h>

// NPerTokenSwishGLU via coefficient-folded GEMMs on gfx1250 WMMA f16 (f32 acc).
// u_unnorm[(b,l),f] = sum_{k,d} (c1[l,k]*x[b,l,d]) * Bg[k,d,f]   -> 2048x8192x1024
// v_unnorm likewise; out_unnorm[(b,l),d] = sum_{k,f} (c3[l,k]*h)*Bd[k,f,d] -> 2048x16384x512
// Column norms of blended weights computed analytically via KxK Gram matrices (f32).
// Bases are converted f32->f16 *pre-transposed* ([N][Kd]) so the GEMM B tile is a pure
// contiguous copy, staged with GLOBAL_LOAD_ASYNC_TO_LDS_B128 (ASYNCcnt) overlapped
// with WMMA compute; A tiles are built on the VALU (coef scale / silu fusion).

typedef __attribute__((ext_vector_type(16))) _Float16 v16h;
typedef __attribute__((ext_vector_type(8)))  float    v8f;

#define B_SZ 16
#define L_SZ 128
#define D_SZ 512
#define F_SZ 1024
#define K_SZ 16
#define M_SZ (B_SZ * L_SZ)          /* 2048 */
#define SQRT_D 22.62741699796952f   /* sqrt(512) */

#define MT 128
#define NT 128
#define KT 32
#define LDK 40                      /* padded k-stride in halves (80B, 16B aligned) */

// ---- CDNA5 async global->LDS copy (16B per lane), tracked on ASYNCcnt ----
__device__ __forceinline__ void async_b128(unsigned lds_off, const void* gptr) {
  asm volatile("global_load_async_to_lds_b128 %0, %1, off"
               :: "v"(lds_off), "v"((unsigned long long)(uintptr_t)gptr)
               : "memory");
}
#define ASYNC_WAIT() asm volatile("s_wait_asynccnt 0x0" ::: "memory")

// ---------------- softmax over K for the three coefficient sets ----------------
__global__ void softmax_coef_kernel(const float* __restrict__ t1,
                                    const float* __restrict__ t2,
                                    const float* __restrict__ t3,
                                    float* __restrict__ out) {
  const int which = blockIdx.x;          // 0..2
  const int l = threadIdx.x;             // 0..127
  const float* src = (which == 0) ? t1 : ((which == 1) ? t2 : t3);
  float v[K_SZ];
  float mx = -1e30f;
#pragma unroll
  for (int k = 0; k < K_SZ; ++k) { v[k] = src[l * K_SZ + k]; mx = fmaxf(mx, v[k]); }
  float s = 0.f;
#pragma unroll
  for (int k = 0; k < K_SZ; ++k) { v[k] = __expf(v[k] - mx); s += v[k]; }
  const float inv = 1.f / s;
#pragma unroll
  for (int k = 0; k < K_SZ; ++k) out[(which * L_SZ + l) * K_SZ + k] = v[k] * inv;
}

// ------- f32 [R][C] -> f16 transposed [C][R] (32x32 LDS tiles, both sides coalesced) -------
__global__ void cvt_t_f16_kernel(const float* __restrict__ src,
                                 _Float16* __restrict__ dst, int R, int C) {
  __shared__ _Float16 tile[32][33];
  const int c0 = blockIdx.x * 32;
  const int r0 = blockIdx.y * 32;
  const int tx = threadIdx.x & 31;
  const int ty = threadIdx.x >> 5;       // 0..7
#pragma unroll
  for (int i = 0; i < 32; i += 8)
    tile[ty + i][tx] = (_Float16)src[(size_t)(r0 + ty + i) * C + (c0 + tx)];
  __syncthreads();
#pragma unroll
  for (int i = 0; i < 32; i += 8)
    dst[(size_t)(c0 + ty + i) * R + (r0 + tx)] = tile[tx][ty + i];
}

// ---------------- Gram: G[k][k2][c] = sum_s basis[k][s][c]*basis[k2][s][c] ----------------
__global__ void gram_kernel(const float* __restrict__ basis,
                            float* __restrict__ G, int S, int C) {
  const int idx = blockIdx.x * blockDim.x + threadIdx.x;   // (k, c)
  if (idx >= K_SZ * C) return;
  const int k = idx / C, c = idx % C;
  float acc[K_SZ];
#pragma unroll
  for (int j = 0; j < K_SZ; ++j) acc[j] = 0.f;
  for (int s = 0; s < S; ++s) {
    const float bk = basis[((size_t)k * S + s) * C + c];
#pragma unroll
    for (int j = 0; j < K_SZ; ++j)
      acc[j] = fmaf(bk, basis[((size_t)j * S + s) * C + c], acc[j]);
  }
#pragma unroll
  for (int j = 0; j < K_SZ; ++j) G[((size_t)k * K_SZ + j) * C + c] = acc[j];
}

// ---------------- invnorm[l][c] = 1 / max(sqrt(c^T G[c] c), eps) ----------------
__global__ void invnorm_kernel(const float* __restrict__ G,
                               const float* __restrict__ coef,
                               float* __restrict__ inv, int C) {
  const int idx = blockIdx.x * blockDim.x + threadIdx.x;
  if (idx >= L_SZ * C) return;
  const int l = idx / C, c = idx % C;
  float cf[K_SZ];
#pragma unroll
  for (int k = 0; k < K_SZ; ++k) cf[k] = coef[l * K_SZ + k];
  float q = 0.f;
  for (int k = 0; k < K_SZ; ++k) {
    const float ck = cf[k];
#pragma unroll
    for (int j = 0; j < K_SZ; ++j)
      q = fmaf(ck * cf[j], G[((size_t)k * K_SZ + j) * C + c], q);
  }
  const float n = sqrtf(fmaxf(q, 0.f));
  inv[idx] = 1.f / fmaxf(n, 1e-12f);
}

// ---------------- GEMM1/2: Out[m][f] = invn[l][f] * sum_kd (c[l][kb]*x[m][d]) * basisT[f][kd]
__global__ __launch_bounds__(256)
void gemm_uv_kernel(const float* __restrict__ x,
                    const _Float16* __restrict__ basisT,   // [F][K*D] f16 (transposed)
                    const float* __restrict__ coef,        // [L][K]
                    const float* __restrict__ invn,        // [L][F]
                    float* __restrict__ out)               // [M][F]
{
  __shared__ _Float16 As[2][MT][LDK];
  __shared__ _Float16 Bs[2][NT][LDK];
  const int t  = threadIdx.x;
  const int n0 = blockIdx.x * NT;
  const int m0 = blockIdx.y * MT;

  const int arow = t >> 1;
  const int acg  = (t & 1) << 4;          // 0 or 16
  const int am   = m0 + arow;
  const int al   = am & (L_SZ - 1);

  const int lane = t & 31;
  const int wave = t >> 5;
  const int wm = (wave & 3) * 32;
  const int wn = (wave >> 2) * 64;
  const int g  = lane >> 4;
  const int ln = lane & 15;

  v8f acc[2][4];
#pragma unroll
  for (int i = 0; i < 2; ++i)
#pragma unroll
    for (int j = 0; j < 4; ++j)
#pragma unroll
      for (int r = 0; r < 8; ++r) acc[i][j][r] = 0.f;

  auto stage = [&](int buf, int kt) {
    const int kd0 = kt * KT;
    { // A: scaled x, row-major [m][k], two b128 LDS stores
      const int kb = kd0 >> 9;            // KT divides 512 -> constant per tile
      const int d0 = (kd0 & (D_SZ - 1)) + acg;
      const float c = coef[al * K_SZ + kb];
      const float4* xp = (const float4*)(x + (size_t)am * D_SZ + d0);
      union { _Float16 h[16]; uint4 q[2]; } pk;
#pragma unroll
      for (int i = 0; i < 4; ++i) {
        const float4 v = xp[i];
        pk.h[4*i+0] = (_Float16)(v.x * c);
        pk.h[4*i+1] = (_Float16)(v.y * c);
        pk.h[4*i+2] = (_Float16)(v.z * c);
        pk.h[4*i+3] = (_Float16)(v.w * c);
      }
      uint4* dst = (uint4*)&As[buf][arow][acg];
      dst[0] = pk.q[0]; dst[1] = pk.q[1];
    }
    { // B: pure async copy of transposed basis rows into [n][k] tile
#pragma unroll
      for (int i = 0; i < 2; ++i) {
        const int unit = t + (i << 8);            // 0..511
        const int nrow = unit >> 2;               // 0..127
        const int ch   = (unit & 3) << 3;         // 0,8,16,24 halves
        const _Float16* gp =
            basisT + (size_t)(n0 + nrow) * (size_t)(K_SZ * D_SZ) + (kd0 + ch);
        async_b128((unsigned)(uintptr_t)&Bs[buf][nrow][ch], gp);
      }
    }
  };

  stage(0, 0);
  ASYNC_WAIT();
  __syncthreads();
  const int NKT = (K_SZ * D_SZ) / KT;     // 256
  for (int kt = 0; kt < NKT; ++kt) {
    const int cur = kt & 1;
    if (kt + 1 < NKT) stage(cur ^ 1, kt + 1);
    v16h a[2], b[4];
#pragma unroll
    for (int fm = 0; fm < 2; ++fm) {      // A frag: halves k = 8g..8g+7, 16+8g..16+8g+7
      union { uint4 q[2]; v16h v; } fr;
      const int row = wm + fm * 16 + ln;
      fr.q[0] = *(const uint4*)&As[cur][row][8 * g];
      fr.q[1] = *(const uint4*)&As[cur][row][16 + 8 * g];
      a[fm] = fr.v;
    }
#pragma unroll
    for (int fn = 0; fn < 4; ++fn) {      // B frag: halves k = 8g..8g+7, 16+8g..16+8g+7
      union { uint4 q[2]; v16h v; } fr;
      const int col = wn + fn * 16 + ln;
      fr.q[0] = *(const uint4*)&Bs[cur][col][8 * g];
      fr.q[1] = *(const uint4*)&Bs[cur][col][16 + 8 * g];
      b[fn] = fr.v;
    }
#pragma unroll
    for (int fm = 0; fm < 2; ++fm)
#pragma unroll
      for (int fn = 0; fn < 4; ++fn)
        acc[fm][fn] = __builtin_amdgcn_wmma_f32_16x16x32_f16(
            false, a[fm], false, b[fn], (short)0, acc[fm][fn], false, false);
    ASYNC_WAIT();
    __syncthreads();
  }
#pragma unroll
  for (int fm = 0; fm < 2; ++fm)
#pragma unroll
    for (int fn = 0; fn < 4; ++fn) {
      const int col = n0 + wn + fn * 16 + ln;
#pragma unroll
      for (int r = 0; r < 8; ++r) {
        const int row = m0 + wm + fm * 16 + 8 * g + r;
        const int l = row & (L_SZ - 1);
        out[(size_t)row * F_SZ + col] = acc[fm][fn][r] * invn[l * F_SZ + col];
      }
    }
}

// ---------------- GEMM3: Y[m][d] = invn3[l][d] * sum_kf (c3[l][kb]*h[m][f]) * downT[d][kf]
// h = silu(|vs[f]|*sqrt(D)*V) * (|us[f]|*U)  computed on the fly during A staging.
__global__ __launch_bounds__(256)
void gemm_down_kernel(const float* __restrict__ U, const float* __restrict__ V,
                      const _Float16* __restrict__ downT,  // [D][K*F] f16 (transposed)
                      const float* __restrict__ coef3,     // [L][K]
                      const float* __restrict__ usp,       // [F]
                      const float* __restrict__ vsp,       // [F]
                      const float* __restrict__ invn3,     // [L][D]
                      float* __restrict__ Y)               // [M][D]
{
  __shared__ _Float16 As[2][MT][LDK];
  __shared__ _Float16 Bs[2][NT][LDK];
  const int t  = threadIdx.x;
  const int n0 = blockIdx.x * NT;
  const int m0 = blockIdx.y * MT;

  const int arow = t >> 1;
  const int acg  = (t & 1) << 4;
  const int am   = m0 + arow;
  const int al   = am & (L_SZ - 1);

  const int lane = t & 31;
  const int wave = t >> 5;
  const int wm = (wave & 3) * 32;
  const int wn = (wave >> 2) * 64;
  const int g  = lane >> 4;
  const int ln = lane & 15;

  v8f acc[2][4];
#pragma unroll
  for (int i = 0; i < 2; ++i)
#pragma unroll
    for (int j = 0; j < 4; ++j)
#pragma unroll
      for (int r = 0; r < 8; ++r) acc[i][j][r] = 0.f;

  auto stage = [&](int buf, int kt) {
    const int kd0 = kt * KT;
    { // A: c3 * silu-gated hidden, computed from U/V
      const int kb = kd0 >> 10;           // KT divides 1024 -> constant per tile
      const int f0 = (kd0 & (F_SZ - 1)) + acg;
      const float c = coef3[al * K_SZ + kb];
      const float4* up = (const float4*)(U + (size_t)am * F_SZ + f0);
      const float4* vp = (const float4*)(V + (size_t)am * F_SZ + f0);
      const float4* sp = (const float4*)(usp + f0);
      const float4* tp = (const float4*)(vsp + f0);
      union { _Float16 h[16]; uint4 q[2]; } pk;
#pragma unroll
      for (int i = 0; i < 4; ++i) {
        const float4 u4 = up[i], v4 = vp[i], s4 = sp[i], t4 = tp[i];
        const float uu[4] = { u4.x * fabsf(s4.x), u4.y * fabsf(s4.y),
                              u4.z * fabsf(s4.z), u4.w * fabsf(s4.w) };
        const float vv[4] = { v4.x * fabsf(t4.x) * SQRT_D, v4.y * fabsf(t4.y) * SQRT_D,
                              v4.z * fabsf(t4.z) * SQRT_D, v4.w * fabsf(t4.w) * SQRT_D };
#pragma unroll
        for (int j = 0; j < 4; ++j) {
          const float sil = vv[j] / (1.f + __expf(-vv[j]));
          pk.h[4*i + j] = (_Float16)(c * sil * uu[j]);
        }
      }
      uint4* dst = (uint4*)&As[buf][arow][acg];
      dst[0] = pk.q[0]; dst[1] = pk.q[1];
    }
    { // B: pure async copy of transposed down-basis rows into [n][k] tile
#pragma unroll
      for (int i = 0; i < 2; ++i) {
        const int unit = t + (i << 8);
        const int nrow = unit >> 2;
        const int ch   = (unit & 3) << 3;
        const _Float16* gp =
            downT + (size_t)(n0 + nrow) * (size_t)(K_SZ * F_SZ) + (kd0 + ch);
        async_b128((unsigned)(uintptr_t)&Bs[buf][nrow][ch], gp);
      }
    }
  };

  stage(0, 0);
  ASYNC_WAIT();
  __syncthreads();
  const int NKT = (K_SZ * F_SZ) / KT;     // 512
  for (int kt = 0; kt < NKT; ++kt) {
    const int cur = kt & 1;
    if (kt + 1 < NKT) stage(cur ^ 1, kt + 1);
    v16h a[2], b[4];
#pragma unroll
    for (int fm = 0; fm < 2; ++fm) {
      union { uint4 q[2]; v16h v; } fr;
      const int row = wm + fm * 16 + ln;
      fr.q[0] = *(const uint4*)&As[cur][row][8 * g];
      fr.q[1] = *(const uint4*)&As[cur][row][16 + 8 * g];
      a[fm] = fr.v;
    }
#pragma unroll
    for (int fn = 0; fn < 4; ++fn) {
      union { uint4 q[2]; v16h v; } fr;
      const int col = wn + fn * 16 + ln;
      fr.q[0] = *(const uint4*)&Bs[cur][col][8 * g];
      fr.q[1] = *(const uint4*)&Bs[cur][col][16 + 8 * g];
      b[fn] = fr.v;
    }
#pragma unroll
    for (int fm = 0; fm < 2; ++fm)
#pragma unroll
      for (int fn = 0; fn < 4; ++fn)
        acc[fm][fn] = __builtin_amdgcn_wmma_f32_16x16x32_f16(
            false, a[fm], false, b[fn], (short)0, acc[fm][fn], false, false);
    ASYNC_WAIT();
    __syncthreads();
  }
#pragma unroll
  for (int fm = 0; fm < 2; ++fm)
#pragma unroll
    for (int fn = 0; fn < 4; ++fn) {
      const int col = n0 + wn + fn * 16 + ln;
#pragma unroll
      for (int r = 0; r < 8; ++r) {
        const int row = m0 + wm + fm * 16 + 8 * g + r;
        const int l = row & (L_SZ - 1);
        Y[(size_t)row * D_SZ + col] = acc[fm][fn][r] * invn3[l * D_SZ + col];
      }
    }
}

// ---------------- final row-wise l2 normalization over D=512 ----------------
__global__ void l2norm_out_kernel(const float* __restrict__ Y, float* __restrict__ out) {
  __shared__ float red[256];
  const int m = blockIdx.x;
  const int t = threadIdx.x;
  const float* y = Y + (size_t)m * D_SZ;
  const float a = y[t], b = y[t + 256];
  red[t] = a * a + b * b;
  __syncthreads();
  for (int off = 128; off > 0; off >>= 1) {
    if (t < off) red[t] += red[t + off];
    __syncthreads();
  }
  const float inv = 1.f / fmaxf(sqrtf(red[0]), 1e-12f);
  out[(size_t)m * D_SZ + t]       = a * inv;
  out[(size_t)m * D_SZ + t + 256] = b * inv;
}

extern "C" void kernel_launch(void* const* d_in, const int* in_sizes, int n_in,
                              void* d_out, int out_size, void* d_ws, size_t ws_size,
                              hipStream_t stream) {
  const float* x   = (const float*)d_in[0];
  const float* Wg  = (const float*)d_in[1];
  const float* Wu  = (const float*)d_in[2];
  const float* Wd  = (const float*)d_in[3];
  const float* tc1 = (const float*)d_in[4];
  const float* tc2 = (const float*)d_in[5];
  const float* tc3 = (const float*)d_in[6];
  const float* usp = (const float*)d_in[7];
  const float* vsp = (const float*)d_in[8];
  (void)in_sizes; (void)n_in; (void)out_size; (void)ws_size;

  char* ws = (char*)d_ws;
  size_t off = 0;
  auto take = [&](size_t bytes) -> char* {
    char* p = ws + off;
    off = (off + bytes + 255) & ~(size_t)255;
    return p;
  };
  const size_t NB = (size_t)K_SZ * D_SZ * F_SZ;      // 8388608 per basis
  float*    cS   = (float*)take(3 * L_SZ * K_SZ * sizeof(float));
  _Float16* WgT  = (_Float16*)take(NB * sizeof(_Float16));  // [F][K*D]
  _Float16* WuT  = (_Float16*)take(NB * sizeof(_Float16));  // [F][K*D]
  _Float16* WdT  = (_Float16*)take(NB * sizeof(_Float16));  // [D][K*F]
  float*    G1   = (float*)take((size_t)K_SZ * K_SZ * F_SZ * sizeof(float));
  float*    G2   = (float*)take((size_t)K_SZ * K_SZ * F_SZ * sizeof(float));
  float*    G3   = (float*)take((size_t)K_SZ * K_SZ * D_SZ * sizeof(float));
  float*    inv1 = (float*)take((size_t)L_SZ * F_SZ * sizeof(float));
  float*    inv2 = (float*)take((size_t)L_SZ * F_SZ * sizeof(float));
  float*    inv3 = (float*)take((size_t)L_SZ * D_SZ * sizeof(float));
  float*    Ubuf = (float*)take((size_t)M_SZ * F_SZ * sizeof(float));
  float*    Vbuf = (float*)take((size_t)M_SZ * F_SZ * sizeof(float));
  float*    Ybuf = (float*)take((size_t)M_SZ * D_SZ * sizeof(float));

  softmax_coef_kernel<<<3, L_SZ, 0, stream>>>(tc1, tc2, tc3, cS);

  // transpose+convert: [K*D][F] -> [F][K*D] and [K*F][D] -> [D][K*F]
  dim3 tguv(F_SZ / 32, (K_SZ * D_SZ) / 32);   // (32, 256)
  cvt_t_f16_kernel<<<tguv, 256, 0, stream>>>(Wg, WgT, K_SZ * D_SZ, F_SZ);
  cvt_t_f16_kernel<<<tguv, 256, 0, stream>>>(Wu, WuT, K_SZ * D_SZ, F_SZ);
  dim3 tgd(D_SZ / 32, (K_SZ * F_SZ) / 32);    // (16, 512)
  cvt_t_f16_kernel<<<tgd, 256, 0, stream>>>(Wd, WdT, K_SZ * F_SZ, D_SZ);

  gram_kernel<<<(K_SZ * F_SZ) / 256, 256, 0, stream>>>(Wg, G1, D_SZ, F_SZ);
  gram_kernel<<<(K_SZ * F_SZ) / 256, 256, 0, stream>>>(Wu, G2, D_SZ, F_SZ);
  gram_kernel<<<(K_SZ * D_SZ) / 256, 256, 0, stream>>>(Wd, G3, F_SZ, D_SZ);

  const float* c1 = cS;
  const float* c2 = cS + L_SZ * K_SZ;
  const float* c3 = cS + 2 * L_SZ * K_SZ;
  invnorm_kernel<<<(L_SZ * F_SZ) / 256, 256, 0, stream>>>(G1, c1, inv1, F_SZ);
  invnorm_kernel<<<(L_SZ * F_SZ) / 256, 256, 0, stream>>>(G2, c2, inv2, F_SZ);
  invnorm_kernel<<<(L_SZ * D_SZ) / 256, 256, 0, stream>>>(G3, c3, inv3, D_SZ);

  dim3 guv(F_SZ / NT, M_SZ / MT);   // (8, 16)
  gemm_uv_kernel<<<guv, 256, 0, stream>>>(x, WgT, c1, inv1, Ubuf);
  gemm_uv_kernel<<<guv, 256, 0, stream>>>(x, WuT, c2, inv2, Vbuf);

  dim3 gdn(D_SZ / NT, M_SZ / MT);   // (4, 16)
  gemm_down_kernel<<<gdn, 256, 0, stream>>>(Ubuf, Vbuf, WdT, c3, usp, vsp, inv3, Ybuf);

  l2norm_out_kernel<<<M_SZ, 256, 0, stream>>>(Ybuf, (float*)d_out);
}
